// KANLayer_original_90486370993053
// MI455X (gfx1250) — matrix operand order repaired and checked by reading, not compiled, not run on hardware
//
#include <hip/hip_runtime.h>

typedef __attribute__((ext_vector_type(16))) _Float16 v16h;
typedef __attribute__((ext_vector_type(8)))  _Float16 v8h;
typedef __attribute__((ext_vector_type(8)))  float    v8f;

#define IN_DIM  256
#define OUT_DIM 256
#define NGRID   12   // NUM+1 + 2*K knots after extension
#define NG      8    // NUM+K spline basis functions per channel
#define TILE_M  64
#define TILE_N  64

// Fused KAN layer: features (silu + cubic B-spline basis on a uniform grid)
// computed on the fly, staged in LDS directly in WMMA fragment layout,
// contracted with v_wmma_f32_16x16x32_f16.
__global__ __launch_bounds__(256) void kan_fused_wmma(
    const float* __restrict__ x,          // (2048, 256)
    const float* __restrict__ grid_,      // (256, 12) -- uniform per row
    const float* __restrict__ coef,       // (256, 256, 8)
    const float* __restrict__ scale_base, // (256, 256)
    const float* __restrict__ scale_sp,   // (256, 256)
    const float* __restrict__ mask,       // (256, 256)
    float* __restrict__ out,              // (2048, 256)
    int batch)
{
    // Fragment-layout LDS staging (contiguous k = 16*(lane/16) + elem, same for A and B):
    //  A (16x32 f16): lanes 0-15 row=lane, elems 0..15 = K0..15;
    //                 lanes 16-31 same row, K16..31.
    //  B (32x16 f16): lanes 0-15 col=lane, elems 0..15 = K0..15;
    //                 lanes 16-31 same col, K16..31.
    __shared__ __attribute__((aligned(32))) _Float16 AsF[4][32][16]; // 4 row-blocks of 16
    __shared__ __attribute__((aligned(32))) _Float16 BsF[4][32][16]; // 4 col-blocks of 16

    const int tid  = threadIdx.x;
    const int lane = tid & 31;
    const int w    = tid >> 5;   // 8 waves
    const int wr   = w & 1;      // row half (32 rows each)
    const int wc   = w >> 1;     // col block (16 cols each)
    const int row0 = blockIdx.x * TILE_M;
    const int col0 = blockIdx.y * TILE_N;

    v8f acc0 = {};
    v8f acc1 = {};

    // ---------------- Phase 1: base path, K = IN_DIM (silu(x) @ scale_base*mask)
    for (int kc = 0; kc < IN_DIM; kc += 32) {
        __syncthreads();
        {
            // A fill: 64 rows x 32 k; 8 consecutive k per thread -> one b128 store
            int r = tid >> 2;
            int q = tid & 3;               // k chunk: k = q*8 + j
            const float* xp = &x[(size_t)(row0 + r) * IN_DIM + kc + q * 8];
            v8h af;
            #pragma unroll
            for (int j = 0; j < 8; ++j) {
                float xv = xp[j];
                // silu = x * sigmoid(x); v_rcp_f32 instead of IEEE divide
                af[j] = (_Float16)(xv * __builtin_amdgcn_rcpf(1.0f + __expf(-xv)));
            }
            *(v8h*)&AsF[r >> 4][(r & 15) | ((q >> 1) << 4)][(q & 1) * 8] = af;

            // B fill: 32 k x 64 cols; fixed k, 8 consecutive cols per thread
            // -> coalesced float4 loads (global_load_b128), scattered b16 stores
            int k2 = tid >> 3;             // 0..31
            int cb = (tid & 7) * 8;        // col chunk
            const float4* bp = (const float4*)&scale_base[(size_t)(kc + k2) * OUT_DIM + col0 + cb];
            const float4* mp = (const float4*)&mask      [(size_t)(kc + k2) * OUT_DIM + col0 + cb];
            float4 b0 = bp[0], b1 = bp[1];
            float4 m0 = mp[0], m1 = mp[1];
            float wv[8] = { b0.x * m0.x, b0.y * m0.y, b0.z * m0.z, b0.w * m0.w,
                            b1.x * m1.x, b1.y * m1.y, b1.z * m1.z, b1.w * m1.w };
            #pragma unroll
            for (int j = 0; j < 8; ++j) {
                int c = cb + j;
                BsF[c >> 4][(c & 15) | ((k2 >> 4) << 4)][k2 & 15] = (_Float16)wv[j];
            }
        }
        __syncthreads();
        v16h a0 = *(const v16h*)&AsF[wr * 2 + 0][lane][0];
        v16h a1 = *(const v16h*)&AsF[wr * 2 + 1][lane][0];
        v16h b  = *(const v16h*)&BsF[wc][lane][0];
        acc0 = __builtin_amdgcn_wmma_f32_16x16x32_f16(false, a0, false, b,
                                                      (short)0, acc0, false, false);
        acc1 = __builtin_amdgcn_wmma_f32_16x16x32_f16(false, a1, false, b,
                                                      (short)0, acc1, false, false);
    }

    // ---------------- Phase 2: spline path, K = IN_DIM*NG
    // 4 input channels per k-step (4*8 = 32 = WMMA K)
    for (int i0 = 0; i0 < IN_DIM; i0 += 4) {
        __syncthreads();
        if (i0 + 4 < IN_DIM)   // uniform branch: prefetch next coef chunk into L2
            __builtin_prefetch(&coef[((size_t)(i0 + 4) * OUT_DIM + col0) * NG], 0, 1);
        {
            // A fill: one (row, channel) basis per thread -> one b128 store.
            // Uniform grid: knot coordinate u = (x - g0)/h; all Cox-de Boor
            // denominators become p*h -> multiply by compile-time 1/p.
            int   r  = tid >> 2;
            int   ci = tid & 3;
            int   i  = i0 + ci;
            float xv = x[(size_t)(row0 + r) * IN_DIM + i];
            const float* gp = &grid_[(size_t)i * NGRID];
            float g0   = gp[0];
            float invh = __builtin_amdgcn_rcpf(gp[1] - g0);
            float u    = (xv - g0) * invh;           // in knot units, [0, 11)
            float fl   = floorf(u);                  // active knot interval
            float B[NGRID - 1];
            #pragma unroll
            for (int t = 0; t < NGRID - 1; ++t)
                B[t] = (fl == (float)t) ? 1.0f : 0.0f;
            #pragma unroll
            for (int p = 1; p <= 3; ++p) {
                const float invp = 1.0f / (float)p;  // constant-folded
                #pragma unroll
                for (int t = 0; t + p < NGRID - 1; ++t) {
                    float l  = (u - (float)t) * invp;
                    float rr = ((float)(t + p + 1) - u) * invp;
                    B[t] = l * B[t] + rr * B[t + 1];
                }
            }
            v8h af;
            #pragma unroll
            for (int gg = 0; gg < NG; ++gg) af[gg] = (_Float16)B[gg];
            // k = ci*8+gg -> lane_hi = ci>>1, elem base = (ci&1)*8 (contiguous)
            *(v8h*)&AsF[r >> 4][(r & 15) | ((ci >> 1) << 4)][(ci & 1) * 8] = af;

            // B fill: one (channel, out-col) per thread -> 8 consecutive coef
            // loads (2x global_load_b128), one b128 store.
            int ci2 = tid >> 6;
            int c   = tid & 63;
            int i2  = i0 + ci2;
            int o   = col0 + c;
            float sm = scale_sp[(size_t)i2 * OUT_DIM + o] * mask[(size_t)i2 * OUT_DIM + o];
            const float* cp = &coef[((size_t)i2 * OUT_DIM + o) * NG];
            v8h bf;
            #pragma unroll
            for (int gg = 0; gg < NG; ++gg)
                bf[gg] = (_Float16)(cp[gg] * sm);
            // k = ci2*8+gg -> lane_hi = ci2>>1, elem base = (ci2&1)*8 (contiguous)
            *(v8h*)&BsF[c >> 4][(c & 15) | ((ci2 >> 1) << 4)][(ci2 & 1) * 8] = bf;
        }
        __syncthreads();
        v16h a0 = *(const v16h*)&AsF[wr * 2 + 0][lane][0];
        v16h a1 = *(const v16h*)&AsF[wr * 2 + 1][lane][0];
        v16h b  = *(const v16h*)&BsF[wc][lane][0];
        acc0 = __builtin_amdgcn_wmma_f32_16x16x32_f16(false, a0, false, b,
                                                      (short)0, acc0, false, false);
        acc1 = __builtin_amdgcn_wmma_f32_16x16x32_f16(false, a1, false, b,
                                                      (short)0, acc1, false, false);
    }

    // ---------------- Epilogue: C/D layout -> global f32 stores
    const int colw  = col0 + wc * 16 + (lane & 15);
    const int rbase = row0 + wr * 32 + (lane >> 4) * 8;
    #pragma unroll
    for (int v = 0; v < 8; ++v) {
        out[(size_t)(rbase + v)      * OUT_DIM + colw] = acc0[v];
        out[(size_t)(rbase + v + 16) * OUT_DIM + colw] = acc1[v];
    }
}

extern "C" void kernel_launch(void* const* d_in, const int* in_sizes, int n_in,
                              void* d_out, int out_size, void* d_ws, size_t ws_size,
                              hipStream_t stream) {
    const float* x    = (const float*)d_in[0];
    const float* grid = (const float*)d_in[1];
    const float* coef = (const float*)d_in[2];
    const float* sb   = (const float*)d_in[3];
    const float* sp   = (const float*)d_in[4];
    const float* mk   = (const float*)d_in[5];
    float* out = (float*)d_out;

    const int batch = in_sizes[0] / IN_DIM;   // 2048
    dim3 grd(batch / TILE_M, OUT_DIM / TILE_N);
    kan_fused_wmma<<<grd, 256, 0, stream>>>(x, grid, coef, sb, sp, mk, out, batch);
}
